// RGCN_79388175499709
// MI455X (gfx1250) — matrix operand order repaired and checked by reading, not compiled, not run on hardware
//
#include <hip/hip_runtime.h>
#include <hip/hip_bf16.h>

typedef __attribute__((ext_vector_type(16))) _Float16 v16h;
typedef __attribute__((ext_vector_type(8)))  _Float16 v8h;
typedef __attribute__((ext_vector_type(4)))  _Float16 v4h;
typedef __attribute__((ext_vector_type(8)))  float    v8f;

#define N_NODES 50000
#define D_IN    300
#define K_PAD   320
#define D_H     128
#define D_OUT   64

// ---------------- conversion / utility kernels ----------------

__global__ void zero_f32(float* __restrict__ p, int n) {
    int i = blockIdx.x * blockDim.x + threadIdx.x;
    if (i < n) p[i] = 0.0f;
}

// x16[m][k] = f16(x[m][k]) for k<300 else 0   (N x 320)
__global__ void cvt_x16(const float* __restrict__ x, _Float16* __restrict__ x16, int total) {
    int i = blockIdx.x * blockDim.x + threadIdx.x;
    if (i >= total) return;
    int m = i / K_PAD, k = i - m * K_PAD;
    x16[i] = (k < D_IN) ? (_Float16)x[m * D_IN + k] : (_Float16)0.0f;
}

// embT[n][k] = f16(emb_w[k][n]) padded  (128 x 320)
__global__ void cvt_embT(const float* __restrict__ w, _Float16* __restrict__ wt) {
    int i = blockIdx.x * blockDim.x + threadIdx.x;
    if (i >= D_H * K_PAD) return;
    int n = i / K_PAD, k = i - n * K_PAD;
    wt[i] = (k < D_IN) ? (_Float16)w[k * D_H + n] : (_Float16)0.0f;
}

// wt[n][k] = f16(w[k][n])  (generic transpose, K x Ncol source)
__global__ void cvt_wT(const float* __restrict__ w, _Float16* __restrict__ wt,
                       int K, int Ncol) {
    int i = blockIdx.x * blockDim.x + threadIdx.x;
    if (i >= K * Ncol) return;
    int n = i / K, k = i - n * K;
    wt[i] = (_Float16)w[k * Ncol + n];
}

__global__ void deg_count(const int* __restrict__ dst, float* __restrict__ deg, int E) {
    int e = blockIdx.x * blockDim.x + threadIdx.x;
    if (e < E) atomicAdd(&deg[dst[e]], 1.0f);
}

__global__ void deg_to_dis(float* __restrict__ d, int n) {
    int i = blockIdx.x * blockDim.x + threadIdx.x;
    if (i < n) d[i] = rsqrtf(d[i] + 1.0f);   // deg includes self loop
}

// out[idx] += hw[idx]*dis[n]^2 + b[c]   (self-loop term + per-relation bias)
__global__ void self_add(const float* __restrict__ hw, const float* __restrict__ dis,
                         const float* __restrict__ b, float* __restrict__ out) {
    int idx = blockIdx.x * blockDim.x + threadIdx.x;
    if (idx >= N_NODES * D_H) return;
    int n = idx >> 7, c = idx & (D_H - 1);
    float di = dis[n];
    out[idx] += hw[idx] * di * di + b[c];
}

// one wave per edge: gather float4 row of hw[src], scale, atomic-add to out[dst]
__global__ void scatter_edges(const int* __restrict__ src, const int* __restrict__ dst,
                              const float* __restrict__ dis, const float* __restrict__ hw,
                              float* __restrict__ out, int E) {
    int wid  = (blockIdx.x * blockDim.x + threadIdx.x) >> 5;
    int lane = threadIdx.x & 31;
    if (wid >= E) return;
    int s = src[wid];
    int d = dst[wid];
    float coef = dis[s] * dis[d];
    const float4 v = *(const float4*)(hw + s * D_H + lane * 4);
    float* o = out + d * D_H + lane * 4;
    atomicAdd(o + 0, v.x * coef);
    atomicAdd(o + 1, v.y * coef);
    atomicAdd(o + 2, v.z * coef);
    atomicAdd(o + 3, v.w * coef);
}

// h32/h16 = relu(in), 4 elements per thread
__global__ void relu_cvt4(const float* __restrict__ in, float* __restrict__ h32,
                          _Float16* __restrict__ h16, int n4) {
    int i = blockIdx.x * blockDim.x + threadIdx.x;
    if (i >= n4) return;
    float4 v = ((const float4*)in)[i];
    v.x = v.x > 0.0f ? v.x : 0.0f;
    v.y = v.y > 0.0f ? v.y : 0.0f;
    v.z = v.z > 0.0f ? v.z : 0.0f;
    v.w = v.w > 0.0f ? v.w : 0.0f;
    ((float4*)h32)[i] = v;
    v4h h = {(_Float16)v.x, (_Float16)v.y, (_Float16)v.z, (_Float16)v.w};
    ((v4h*)h16)[i] = h;
}

// ---------------- WMMA GEMM kernels ----------------
// A layout (16x32 f16): lane l -> row l&15 ; half h -> K = (h&8)*2 + 8*(l>>4) + (h&7)
//   => halves 0-7 contiguous at K=kt+8g, halves 8-15 contiguous at K=kt+16+8g
// B layout (32x16 f16): lane l -> col l&15 ; half h -> K = (h&8)*2 + 8*(l>>4) + (h&7)
//   => halves 0-7 at K=kt+8g, halves 8-15 at K=kt+16+8g (column-major weight rows)
// C/D layout: VGPR v -> row v + 8*(l>>4), col l&15

__device__ __forceinline__ v16h ld_a(const _Float16* __restrict__ row, int kt, int g) {
    union { v16h v; v8h h[2]; } u;
    u.h[0] = *(const v8h*)(row + kt + (g << 3));
    u.h[1] = *(const v8h*)(row + kt + 16 + (g << 3));
    return u.v;
}

__device__ __forceinline__ v16h ld_b(const _Float16* __restrict__ col, int kt, int g) {
    union { v16h v; v8h h[2]; } u;
    u.h[0] = *(const v8h*)(col + kt + (g << 3));
    u.h[1] = *(const v8h*)(col + kt + 16 + (g << 3));
    return u.v;
}

// h = relu(x16 @ embT^T + emb_b) ; K padded to 320, 8 waves = 8 col tiles
__global__ void emb_gemm(const _Float16* __restrict__ x16, const _Float16* __restrict__ wT,
                         const float* __restrict__ b, float* __restrict__ h32,
                         _Float16* __restrict__ h16) {
    int lane = threadIdx.x & 31;
    int wave = threadIdx.x >> 5;
    int row0 = blockIdx.x * 16;
    int col0 = wave * 16;
    int g  = lane >> 4;
    int lm = lane & 15;
    const _Float16* arow = x16 + (size_t)(row0 + lm) * K_PAD;
    const _Float16* bcol = wT + (size_t)(col0 + lm) * K_PAD;
    v8f acc = {};
#pragma unroll
    for (int kt = 0; kt < K_PAD; kt += 32) {
        v16h a  = ld_a(arow, kt, g);
        v16h bb = ld_b(bcol, kt, g);
        acc = __builtin_amdgcn_wmma_f32_16x16x32_f16(false, a, false, bb,
                                                     (short)0, acc, false, false);
    }
#pragma unroll
    for (int v = 0; v < 8; ++v) {
        int m = row0 + v + (g << 3);
        int n = col0 + lm;
        float val = acc[v] + b[n];
        val = val > 0.0f ? val : 0.0f;
        h32[m * D_H + n] = val;
        h16[m * D_H + n] = (_Float16)val;
    }
}

// hw = h16 @ wT^T  (128x128, f16 column-major weights)
__global__ void layer_gemm(const _Float16* __restrict__ h16, const _Float16* __restrict__ wT,
                           float* __restrict__ hw) {
    int lane = threadIdx.x & 31;
    int wave = threadIdx.x >> 5;
    int row0 = blockIdx.x * 16;
    int col0 = wave * 16;
    int g  = lane >> 4;
    int lm = lane & 15;
    const _Float16* arow = h16 + (size_t)(row0 + lm) * D_H;
    const _Float16* bcol = wT + (size_t)(col0 + lm) * D_H;
    v8f acc = {};
#pragma unroll
    for (int kt = 0; kt < D_H; kt += 32) {
        v16h a  = ld_a(arow, kt, g);
        v16h bb = ld_b(bcol, kt, g);
        acc = __builtin_amdgcn_wmma_f32_16x16x32_f16(false, a, false, bb,
                                                     (short)0, acc, false, false);
    }
#pragma unroll
    for (int v = 0; v < 8; ++v)
        hw[(row0 + v + (g << 3)) * D_H + col0 + lm] = acc[v];
}

// out = h16 @ linT^T + lin_b  ([N,128]@[128,64]) ; 4 waves per block
__global__ void final_gemm(const _Float16* __restrict__ h16, const _Float16* __restrict__ wT,
                           const float* __restrict__ b, float* __restrict__ out) {
    int lane = threadIdx.x & 31;
    int wave = threadIdx.x >> 5;
    int row0 = blockIdx.x * 16;
    int col0 = wave * 16;
    int g  = lane >> 4;
    int lm = lane & 15;
    const _Float16* arow = h16 + (size_t)(row0 + lm) * D_H;
    const _Float16* bcol = wT + (size_t)(col0 + lm) * D_H;
    v8f acc = {};
#pragma unroll
    for (int kt = 0; kt < D_H; kt += 32) {
        v16h a  = ld_a(arow, kt, g);
        v16h bb = ld_b(bcol, kt, g);
        acc = __builtin_amdgcn_wmma_f32_16x16x32_f16(false, a, false, bb,
                                                     (short)0, acc, false, false);
    }
#pragma unroll
    for (int v = 0; v < 8; ++v) {
        int m = row0 + v + (g << 3);
        int n = col0 + lm;
        out[m * D_OUT + n] = acc[v] + b[n];
    }
}

// ---------------- launch ----------------

extern "C" void kernel_launch(void* const* d_in, const int* in_sizes, int n_in,
                              void* d_out, int out_size, void* d_ws, size_t ws_size,
                              hipStream_t stream) {
    const float* x     = (const float*)d_in[0];
    const int*   ei    = (const int*)d_in[1];     // [3,2,E] int32
    const float* emb_w = (const float*)d_in[2];
    const float* emb_b = (const float*)d_in[3];
    const float* w0    = (const float*)d_in[4];   // [3,128,128]
    const float* b0    = (const float*)d_in[5];   // [3,128]
    const float* w1    = (const float*)d_in[6];
    const float* b1    = (const float*)d_in[7];
    const float* lin_w = (const float*)d_in[8];
    const float* lin_b = (const float*)d_in[9];
    float* out = (float*)d_out;

    const int E = in_sizes[1] / 6;                // edges per relation

    // workspace partition (256B aligned)
    char* ws = (char*)d_ws;
    size_t off = 0;
    auto alloc = [&](size_t bytes) { void* p = ws + off; off = (off + bytes + 255) & ~255ULL; return p; };
    float*    h32  = (float*)   alloc((size_t)N_NODES * D_H * 4);
    _Float16* h16  = (_Float16*)alloc((size_t)N_NODES * D_H * 2);
    float*    hw   = (float*)   alloc((size_t)N_NODES * D_H * 4);
    float*    acc  = (float*)   alloc((size_t)N_NODES * D_H * 4);
    float*    dis  = (float*)   alloc((size_t)3 * N_NODES * 4);
    _Float16* x16  = (_Float16*)alloc((size_t)N_NODES * K_PAD * 2);
    _Float16* embT = (_Float16*)alloc((size_t)D_H * K_PAD * 2);
    _Float16* wT0  = (_Float16*)alloc((size_t)3 * D_H * D_H * 2);
    _Float16* wT1  = (_Float16*)alloc((size_t)3 * D_H * D_H * 2);
    _Float16* linT = (_Float16*)alloc((size_t)D_OUT * D_H * 2);

    const int NT = 256;
    const int NH = N_NODES * D_H;

    // ---- one-time conversions (inputs are static per launch) ----
    {
        int tx = N_NODES * K_PAD;
        cvt_x16<<<(tx + NT - 1) / NT, NT, 0, stream>>>(x, x16, tx);
        cvt_embT<<<(D_H * K_PAD + NT - 1) / NT, NT, 0, stream>>>(emb_w, embT);
        for (int r = 0; r < 3; ++r) {
            cvt_wT<<<(D_H * D_H + NT - 1) / NT, NT, 0, stream>>>(w0 + r * D_H * D_H,
                                                                 wT0 + r * D_H * D_H, D_H, D_H);
            cvt_wT<<<(D_H * D_H + NT - 1) / NT, NT, 0, stream>>>(w1 + r * D_H * D_H,
                                                                 wT1 + r * D_H * D_H, D_H, D_H);
        }
        cvt_wT<<<(D_H * D_OUT + NT - 1) / NT, NT, 0, stream>>>(lin_w, linT, D_H, D_OUT);
    }

    // ---- degree -> D^{-1/2} per relation (edges static across layers) ----
    zero_f32<<<(3 * N_NODES + NT - 1) / NT, NT, 0, stream>>>(dis, 3 * N_NODES);
    for (int r = 0; r < 3; ++r) {
        const int* dst_r = ei + (r * 2 + 1) * E;
        deg_count<<<(E + NT - 1) / NT, NT, 0, stream>>>(dst_r, dis + r * N_NODES, E);
    }
    deg_to_dis<<<(3 * N_NODES + NT - 1) / NT, NT, 0, stream>>>(dis, 3 * N_NODES);

    // ---- embedding ----
    emb_gemm<<<N_NODES / 16, 256, 0, stream>>>(x16, embT, emb_b, h32, h16);

    // ---- two hetero-conv layers ----
    const _Float16* Ws[2] = {wT0, wT1};
    const float*    Bs[2] = {b0, b1};
    for (int layer = 0; layer < 2; ++layer) {
        zero_f32<<<(NH + NT - 1) / NT, NT, 0, stream>>>(acc, NH);
        for (int r = 0; r < 3; ++r) {
            const int* src_r = ei + (r * 2 + 0) * E;
            const int* dst_r = ei + (r * 2 + 1) * E;
            layer_gemm<<<N_NODES / 16, 256, 0, stream>>>(h16, Ws[layer] + r * D_H * D_H, hw);
            self_add<<<(NH + NT - 1) / NT, NT, 0, stream>>>(hw, dis + r * N_NODES,
                                                            Bs[layer] + r * D_H, acc);
            scatter_edges<<<(E * 32 + NT - 1) / NT, NT, 0, stream>>>(src_r, dst_r,
                                                                     dis + r * N_NODES,
                                                                     hw, acc, E);
        }
        relu_cvt4<<<(NH / 4 + NT - 1) / NT, NT, 0, stream>>>(acc, h32, h16, NH / 4);
    }

    // ---- final projection ----
    final_gemm<<<N_NODES / 16, 128, 0, stream>>>(h16, linT, lin_b, out);
}
